// DVGOMoE_11544872091651
// MI455X (gfx1250) — matrix operand order repaired and checked by reading, not compiled, not run on hardware
//
#include <hip/hip_runtime.h>
#include <hip/hip_bf16.h>

#define PTS   1048576
#define RAYS  8192
#define NEXP  8
#define HD    128
#define ACT_SHIFT (-4.0f)
#define INTERVAL   (0.5f)
#define THRESH     (1e-4f)

typedef __attribute__((ext_vector_type(16))) _Float16 v16h;
typedef __attribute__((ext_vector_type(8)))  _Float16 v8h;
typedef __attribute__((ext_vector_type(8)))  float    v8f;

__device__ __forceinline__ float softplus_f(float x) {
    return (x > 20.f) ? x : log1pf(expf(x));
}
__device__ __forceinline__ float raw2alpha_f(float d) {
    return 1.f - expf(-softplus_f(d + ACT_SHIFT) * INTERVAL);
}
__device__ __forceinline__ float sigmoid_f(float x) {
    return 1.f / (1.f + expf(-x));
}

// ---------------------------------------------------------------------------
// Kernel 1: per-point density + gate + 8 experts, all via f16 WMMA.
// Block = 256 threads = 8 waves; each wave owns one 16-point tile at a time,
// 8 tiles per wave per block -> 1024 points per block, 1024 blocks.
// Outputs: wsAlpha0[P] (thresholded density alpha), wsPA4[P][4] = {rgb, alpha_mix}.
// ---------------------------------------------------------------------------
__global__ __launch_bounds__(256) void moe_field_kernel(
    const float* __restrict__ ray_pts, const float* __restrict__ viewdirs,
    const int* __restrict__ ray_id,    const float* __restrict__ Wg,
    const float* __restrict__ Wd1,     const float* __restrict__ bd1,
    const float* __restrict__ Wd2,     const float* __restrict__ We1,
    const float* __restrict__ be1,     const float* __restrict__ Wrgb,
    const float* __restrict__ Walpha,
    float* __restrict__ wsAlpha0,      float* __restrict__ wsPA4)
{
    // slot 0..7 = experts, slot 8 = density MLP (rows 3..5 / cols 1..3 zero)
    __shared__ __align__(16) _Float16 sW1[9][6][HD];   // layer-1 weights, K-major
    __shared__ __align__(16) _Float16 sW2[9][4][HD];   // layer-2 weights, N-major
    __shared__ __align__(16) float sB1[9][HD];
    __shared__ __align__(16) float sWg[48];
    __shared__ __align__(16) _Float16 sH[8][16][HD];   // per-wave hidden staging
    __shared__ float sG[8][16][NEXP];                  // per-wave gate weights

    const int t = threadIdx.x;

    // ---- f32 tables with no conversion: DMA straight to LDS (ASYNCcnt path) ----
    {
        // be1: 8*128 = 1024 floats -> sB1[0..7][*], 256 lanes x 16B
        {
            const float* g = be1 + t * 4;
            unsigned l = (unsigned)(size_t)&sB1[0][0] + (unsigned)t * 16u;
            asm volatile("global_load_async_to_lds_b128 %0, %1, off"
                         :: "v"(l), "v"(g) : "memory");
        }
        if (t < 32) {   // bd1: 128 floats -> sB1[8][*]
            const float* g = bd1 + t * 4;
            unsigned l = (unsigned)(size_t)&sB1[8][0] + (unsigned)t * 16u;
            asm volatile("global_load_async_to_lds_b128 %0, %1, off"
                         :: "v"(l), "v"(g) : "memory");
        }
        if (t < 12) {   // Wg: 48 floats -> sWg
            const float* g = Wg + t * 4;
            unsigned l = (unsigned)(size_t)&sWg[0] + (unsigned)t * 16u;
            asm volatile("global_load_async_to_lds_b128 %0, %1, off"
                         :: "v"(l), "v"(g) : "memory");
        }
    }

    // ---- f16 weight tables (need f32->f16 convert, so load/cvt/ds_store) ----
    for (int idx = t; idx < 9 * 6 * HD; idx += 256) {
        int h = idx & (HD - 1); int k = (idx / HD) % 6; int s = idx / (6 * HD);
        float v;
        if (s < NEXP)   v = We1[(s * 6 + k) * HD + h];
        else if (k < 3) v = Wd1[k * HD + h];
        else            v = 0.f;
        sW1[s][k][h] = (_Float16)v;
    }
    for (int idx = t; idx < 9 * 4 * HD; idx += 256) {
        int h = idx & (HD - 1); int n = (idx / HD) & 3; int s = idx / (4 * HD);
        float v;
        if (s < NEXP) v = (n < 3) ? Wrgb[(s * HD + h) * 3 + n] : Walpha[s * HD + h];
        else          v = (n == 0) ? Wd2[h] : 0.f;
        sW2[s][n][h] = (_Float16)v;
    }
    asm volatile("s_wait_asynccnt 0" ::: "memory");
    __syncthreads();

    const int w      = t >> 5;
    const int lane   = t & 31;
    const int nIdx   = lane & 15;       // column within 16-wide tile
    const int hiHalf = lane >> 4;       // 0 or 1
    const int mOff   = hiHalf << 3;     // row offset for C/D layout

    #pragma unroll 1
    for (int i = 0; i < 8; ++i) {
        const int tile = (blockIdx.x * 8 + w) * 8 + i;
        const int base = tile * 16;

        // prefetch next tile's point stream behind this tile's WMMA work
        __builtin_prefetch(ray_pts + 3 * (base + 16) + 3 * nIdx, 0, 1);
        __builtin_prefetch(ray_id + base + 16 + nIdx, 0, 1);

        // ---- load per-point inputs, build shared A tile (16x32 f16, K=0..5) ----
        float x0 = 0, x1 = 0, x2 = 0, x3 = 0, x4 = 0, x5 = 0;
        if (lane < 16) {
            int p = base + lane;
            x0 = ray_pts[3 * p];     x1 = ray_pts[3 * p + 1]; x2 = ray_pts[3 * p + 2];
            int r = ray_id[p];
            x3 = viewdirs[3 * r];    x4 = viewdirs[3 * r + 1]; x5 = viewdirs[3 * r + 2];
        }
        v16h A;
        #pragma unroll
        for (int j = 0; j < 16; ++j) A[j] = (_Float16)0.f;
        if (lane < 16) {
            A[0] = (_Float16)x0; A[1] = (_Float16)x1; A[2] = (_Float16)x2;
            A[3] = (_Float16)x3; A[4] = (_Float16)x4; A[5] = (_Float16)x5;
        }

        // ---- gate: softmax over 8 experts, top-2, renormalize ----
        if (lane < 16) {
            float lg[NEXP];
            #pragma unroll
            for (int e = 0; e < NEXP; ++e)
                lg[e] = x0 * sWg[e]      + x1 * sWg[8 + e]  + x2 * sWg[16 + e] +
                        x3 * sWg[24 + e] + x4 * sWg[32 + e] + x5 * sWg[40 + e];
            float m = lg[0];
            #pragma unroll
            for (int e = 1; e < NEXP; ++e) m = fmaxf(m, lg[e]);
            float pe[NEXP];
            #pragma unroll
            for (int e = 0; e < NEXP; ++e) pe[e] = expf(lg[e] - m);
            // top-2 (softmax denominator cancels in the renormalization)
            int i1 = 0;
            #pragma unroll
            for (int e = 1; e < NEXP; ++e) if (pe[e] > pe[i1]) i1 = e;
            int i2 = (i1 == 0) ? 1 : 0;
            #pragma unroll
            for (int e = 0; e < NEXP; ++e) if (e != i1 && pe[e] > pe[i2]) i2 = e;
            float denom = pe[i1] + pe[i2];
            #pragma unroll
            for (int e = 0; e < NEXP; ++e)
                sG[w][lane][e] = (e == i1) ? pe[i1] / denom
                               : (e == i2) ? pe[i2] / denom : 0.f;
        }

        float accMix[8];
        #pragma unroll
        for (int v = 0; v < 8; ++v) accMix[v] = 0.f;

        // ---- 9 slots: 8 experts + density, same WMMA pipeline ----
        #pragma unroll 1
        for (int s = 0; s < 9; ++s) {
            // layer 1: x[16x6] @ W1[6x128] + b, ReLU  -> sH (f16)
            #pragma unroll
            for (int c = 0; c < 8; ++c) {
                v16h B;
                #pragma unroll
                for (int j = 0; j < 16; ++j) B[j] = (_Float16)0.f;
                if (lane < 16) {
                    #pragma unroll
                    for (int j = 0; j < 6; ++j) B[j] = sW1[s][j][c * 16 + lane];
                }
                float bias = sB1[s][c * 16 + nIdx];
                v8f C;
                #pragma unroll
                for (int v = 0; v < 8; ++v) C[v] = bias;
                v8f D = __builtin_amdgcn_wmma_f32_16x16x32_f16(
                    false, A, false, B, (short)0, C, false, false);
                #pragma unroll
                for (int v = 0; v < 8; ++v) {
                    float hv = fmaxf(D[v], 0.f);
                    sH[w][v + mOff][c * 16 + nIdx] = (_Float16)hv;
                }
            }
            // wave-local LDS RAW fence before cross-lane reads of sH
            asm volatile("s_wait_dscnt 0" ::: "memory");

            // layer 2: h[16x128] @ W2[128x4-pad-16], K split into 4 chunks of 32
            v8f acc2;
            #pragma unroll
            for (int v = 0; v < 8; ++v) acc2[v] = 0.f;
            const _Float16* hrow = &sH[w][nIdx][0];
            #pragma unroll
            for (int kc = 0; kc < 4; ++kc) {
                v8h lo = *(const v8h*)(hrow + kc * 32 + mOff);
                v8h hi = *(const v8h*)(hrow + kc * 32 + 16 + mOff);
                v16h A2;
                #pragma unroll
                for (int j = 0; j < 8; ++j) { A2[j] = lo[j]; A2[j + 8] = hi[j]; }
                v16h B2;
                #pragma unroll
                for (int j = 0; j < 16; ++j) B2[j] = (_Float16)0.f;
                if (nIdx < 4) {
                    const _Float16* wr = &sW2[s][nIdx][0];
                    const int k0 = kc * 32 + (hiHalf << 3);
                    v8h l2 = *(const v8h*)(wr + k0);
                    v8h h2 = *(const v8h*)(wr + k0 + 16);
                    #pragma unroll
                    for (int j = 0; j < 8; ++j) { B2[j] = l2[j]; B2[j + 8] = h2[j]; }
                }
                acc2 = __builtin_amdgcn_wmma_f32_16x16x32_f16(
                    false, A2, false, B2, (short)0, acc2, false, false);
            }

            if (s < NEXP) {
                // only columns 0..3 are ever stored: mask the transcendental work
                if (nIdx < 4) {
                    #pragma unroll
                    for (int v = 0; v < 8; ++v) {
                        float g = sG[w][v + mOff][s];
                        float o = acc2[v];
                        float val = (nIdx == 3) ? raw2alpha_f(o) : sigmoid_f(o);
                        accMix[v] += g * val;
                    }
                }
            } else if (nIdx == 0) {
                // density slot: column 0 = density -> alpha0 with threshold
                #pragma unroll
                for (int v = 0; v < 8; ++v) {
                    float a0 = raw2alpha_f(acc2[v]);
                    wsAlpha0[base + v + mOff] = (a0 > THRESH) ? a0 : 0.f;
                }
            }
        }

        // ---- write mixed {rgb, alpha_mix} per point ----
        if (nIdx < 4) {
            #pragma unroll
            for (int v = 0; v < 8; ++v)
                wsPA4[(base + v + mOff) * 4 + nIdx] = accMix[v];
        }
    }
}

// ---------------------------------------------------------------------------
// Kernel 2: per-ray sequential compositing (both transmittance scans fused).
// ray_id is sorted; one thread per ray binary-searches its segment.
// ---------------------------------------------------------------------------
__global__ __launch_bounds__(256) void composite_kernel(
    const int* __restrict__ ray_id, const float* __restrict__ wsAlpha0,
    const float* __restrict__ wsPA4, const float* __restrict__ bg,
    float* __restrict__ out)
{
    int r = blockIdx.x * 256 + threadIdx.x;
    if (r >= RAYS) return;

    int lo = 0, hi = PTS;
    while (lo < hi) { int mid = (lo + hi) >> 1; if (ray_id[mid] < r) lo = mid + 1; else hi = mid; }
    int start = lo;
    hi = PTS;
    while (lo < hi) { int mid = (lo + hi) >> 1; if (ray_id[mid] <= r) lo = mid + 1; else hi = mid; }
    int end = lo;

    float T0 = 1.f, T = 1.f, cr = 0.f, cg = 0.f, cb = 0.f;
    for (int j = start; j < end; ++j) {
        float a0 = wsAlpha0[j];
        float w0 = a0 * T0;           // transmittance weight from density pass
        T0 *= (1.f - a0);
        float4 pa = *(const float4*)(wsPA4 + 4 * j);
        float a = (w0 > THRESH) ? pa.w : 0.f;   // keep-mask gates alpha_mix
        float wgt = a * T;
        T *= (1.f - a);
        cr = fmaf(wgt, pa.x, cr);
        cg = fmaf(wgt, pa.y, cg);
        cb = fmaf(wgt, pa.z, cb);
    }
    out[3 * r + 0] = cr + T * bg[0];
    out[3 * r + 1] = cg + T * bg[1];
    out[3 * r + 2] = cb + T * bg[2];
}

extern "C" void kernel_launch(void* const* d_in, const int* in_sizes, int n_in,
                              void* d_out, int out_size, void* d_ws, size_t ws_size,
                              hipStream_t stream) {
    (void)in_sizes; (void)n_in; (void)out_size; (void)ws_size;
    const float* ray_pts  = (const float*)d_in[0];
    const float* viewdirs = (const float*)d_in[1];
    const float* bg       = (const float*)d_in[2];
    const float* Wd1      = (const float*)d_in[3];
    const float* bd1      = (const float*)d_in[4];
    const float* Wd2      = (const float*)d_in[5];
    const float* Wg       = (const float*)d_in[6];
    const float* We1      = (const float*)d_in[7];
    const float* be1      = (const float*)d_in[8];
    const float* Wrgb     = (const float*)d_in[9];
    const float* Walpha   = (const float*)d_in[10];
    const int*   ray_id   = (const int*)d_in[11];

    float* wsAlpha0 = (float*)d_ws;        // P floats
    float* wsPA4    = wsAlpha0 + PTS;      // 4P floats {rgb.xyz, alpha_mix}
    float* out      = (float*)d_out;       // [N,3]

    // 1024 blocks x 256 threads: 8 waves x 8 tiles x 16 points = 1024 pts/block
    moe_field_kernel<<<dim3(PTS / 1024), dim3(256), 0, stream>>>(
        ray_pts, viewdirs, ray_id, Wg, Wd1, bd1, Wd2, We1, be1, Wrgb, Walpha,
        wsAlpha0, wsPA4);

    composite_kernel<<<dim3(RAYS / 256), dim3(256), 0, stream>>>(
        ray_id, wsAlpha0, wsPA4, bg, out);
}
